// TransReID_21337397526674
// MI455X (gfx1250) — matrix operand order repaired and checked by reading, not compiled, not run on hardware
//
#include <hip/hip_runtime.h>
#include <hip/hip_bf16.h>
#include <math.h>
#include <stdint.h>

typedef __bf16 bf16;
typedef __attribute__((ext_vector_type(16))) __bf16 v16bf;
typedef __attribute__((ext_vector_type(8)))  __bf16 v8bf;
typedef __attribute__((ext_vector_type(8)))  float  v8f;

// Model constants
#define DM    768
#define NSEQ  211
#define NB    64
#define NHEAD 12
#define HDIM  64
#define MLPD  3072
#define NPAT  210     // 21 * 10
#define NSEG  53      // 1 + 52
#define NSEGB 256     // 4 segments * 64 batch
#define MMAX  13568   // max GEMM rows (256*53); 64*211=13504, 64*210=13440

__device__ __forceinline__ v8f wmma_bf(v16bf a, v16bf b, v8f c) {
  return __builtin_amdgcn_wmma_f32_16x16x32_bf16(false, a, false, b, (short)0, c,
                                                 false, false);
}

__device__ __forceinline__ float gelu_f(float v) {
  return 0.5f * v * (1.0f + erff(v * 0.70710678118654752f));
}

// CDNA5 async global->LDS copy (16B per lane), tracked by ASYNCcnt.
// Generic shared pointer: addr[31:0] is the LDS byte offset (flat aperture).
__device__ __forceinline__ void async_cp16(void* lds, const void* g) {
  asm volatile("global_load_async_to_lds_b128 %0, %1, off"
               :: "v"((unsigned)(uintptr_t)lds), "v"(g)
               : "memory");
}
__device__ __forceinline__ void wait_async_le2() {
  asm volatile("s_wait_asynccnt 0x2" ::: "memory");
}
__device__ __forceinline__ void wait_async_0() {
  asm volatile("s_wait_asynccnt 0x0" ::: "memory");
}

// ---------------------------------------------------------------- conversions
__global__ __launch_bounds__(256) void k_cvt(const float* __restrict__ s,
                                             bf16* __restrict__ d, long n) {
  long i = (long)blockIdx.x * 256 + threadIdx.x;
  const long stride = (long)gridDim.x * 256;
  for (; i < n; i += stride) d[i] = (bf16)s[i];
}

// conv_w (O=768, K=768) -> WcT (K, O) bf16
__global__ __launch_bounds__(256) void k_cvtT(const float* __restrict__ s,
                                              bf16* __restrict__ d) {
  int o = blockIdx.x * 16 + (threadIdx.x & 15);
  int k = blockIdx.y * 16 + (threadIdx.x >> 4);
  d[(size_t)k * DM + o] = (bf16)s[(size_t)o * DM + k];
}

// ---------------------------------------------------------------- cls row
__global__ __launch_bounds__(256) void k_cls(const float* __restrict__ cls,
                                             const float* __restrict__ pos,
                                             const float* __restrict__ sie,
                                             const int* __restrict__ cids,
                                             float* __restrict__ t) {
  const int b = blockIdx.x, tid = threadIdx.x;
  const int cid = cids[b];
  float* dst = t + (size_t)b * NSEQ * DM;
#pragma unroll
  for (int i = 0; i < 3; ++i) {
    int c = tid + i * 256;
    dst[c] = cls[c] + pos[c] + 3.0f * sie[cid * DM + c];
  }
}

// ---------------------------------------------------------------- generic GEMM
// C(M,N) = A(M,K)bf16 @ W(K,N)bf16 + bias ; EPI: 1=+res->f32, 2=gelu->bf16, 3=->bf16
// Double-buffered LDS staging via async global->LDS copies.
template <int EPI>
__global__ __launch_bounds__(256) void k_gemm(
    const bf16* __restrict__ A, const bf16* __restrict__ W,
    const float* __restrict__ bias, const float* __restrict__ res,
    float* __restrict__ outF, bf16* __restrict__ outB, int M, int K, int Ncols) {
  __shared__ __align__(16) bf16 sA[2][64 * 40];
  __shared__ __align__(16) bf16 sB[2][32 * 72];
  const int tid = threadIdx.x;
  const int lane = tid & 31, wave = tid >> 5;
  const int wr = wave & 3, wc = wave >> 2;
  const int m0 = blockIdx.y * 64, n0 = blockIdx.x * 64;
  v8f acc0 = {}, acc1 = {};
  const int ar = tid >> 2, ac = (tid & 3) << 3;
  const int br = tid >> 3, bc = (tid & 7) << 3;

  // Clamp OOB rows to a valid address; their results are never stored.
  int gra = m0 + ar;
  if (gra >= M) gra = M - 1;
  const bf16* aSrc = A + (size_t)gra * K + ac;
  const bf16* bSrc = W + (size_t)br * Ncols + n0 + bc;
  bf16* aDst0 = &sA[0][ar * 40 + ac];
  bf16* aDst1 = &sA[1][ar * 40 + ac];
  bf16* bDst0 = &sB[0][br * 72 + bc];
  bf16* bDst1 = &sB[1][br * 72 + bc];

  const int nIter = K >> 5;
  async_cp16(aDst0, aSrc);
  async_cp16(bDst0, bSrc);

  for (int it = 0; it < nIter; ++it) {
    const int cur = it & 1;
    if (it + 1 < nIter) {
      async_cp16(cur ? aDst0 : aDst1, aSrc + (size_t)(it + 1) * 32);
      async_cp16(cur ? bDst0 : bDst1, bSrc + (size_t)(it + 1) * 32 * Ncols);
      wait_async_le2();  // current tile's 2 copies complete (in-order)
    } else {
      wait_async_0();
    }
    __syncthreads();
    {
      const bf16* ap = &sA[cur][(wr * 16 + (lane & 15)) * 40 + ((lane >> 4) << 3)];
      v8bf alo = *(const v8bf*)ap;
      v8bf ahi = *(const v8bf*)(ap + 16);
      v16bf af;
#pragma unroll
      for (int i = 0; i < 8; ++i) { af[i] = alo[i]; af[i + 8] = ahi[i]; }
      const int kb = (lane >> 4) << 3;
      const bf16* bp = &sB[cur][wc * 32 + (lane & 15)];
      v16bf bf0, bf1;
#pragma unroll
      for (int i = 0; i < 8; ++i) {
        bf0[i] = bp[(kb + i) * 72]; bf0[i + 8] = bp[(kb + 16 + i) * 72];
        bf1[i] = bp[(kb + i) * 72 + 16]; bf1[i + 8] = bp[(kb + 16 + i) * 72 + 16];
      }
      acc0 = wmma_bf(af, bf0, acc0);
      acc1 = wmma_bf(af, bf1, acc1);
    }
    if (it + 1 < nIter) __syncthreads();  // protect buffer reuse
  }
  const int rbase = m0 + wr * 16 + ((lane >> 4) << 3);
  const int c0 = n0 + wc * 32 + (lane & 15);
#pragma unroll
  for (int j = 0; j < 8; ++j) {
    const int row = rbase + j;
    if (row >= M) break;
    float v0 = acc0[j] + bias[c0];
    float v1 = acc1[j] + bias[c0 + 16];
    const size_t o0 = (size_t)row * Ncols + c0;
    const size_t o1 = o0 + 16;
    if (EPI == 1) {
      outF[o0] = res[o0] + v0;
      outF[o1] = res[o1] + v1;
    } else if (EPI == 2) {
      outB[o0] = (bf16)gelu_f(v0);
      outB[o1] = (bf16)gelu_f(v1);
    } else {
      outB[o0] = (bf16)v0;
      outB[o1] = (bf16)v1;
    }
  }
}

// ---------------------------------------------------------------- patch embed
// t[b, 1+p, :] = im2col(x)[r] @ WcT + conv_b + pos[1+p] + 3*sie[cid]
// (A tile needs f32->bf16 conversion, so it stays on the manual staging path.)
__global__ __launch_bounds__(256) void k_patch_embed(
    const float* __restrict__ x, const bf16* __restrict__ WcT,
    const float* __restrict__ convb, const float* __restrict__ pos,
    const float* __restrict__ sie, const int* __restrict__ cids,
    float* __restrict__ t) {
  __shared__ __align__(16) bf16 sA[64 * 40];
  __shared__ __align__(16) bf16 sB[32 * 72];
  const int tid = threadIdx.x;
  const int lane = tid & 31, wave = tid >> 5;
  const int wr = wave & 3, wc = wave >> 2;
  const int m0 = blockIdx.y * 64, n0 = blockIdx.x * 64;
  v8f acc0 = {}, acc1 = {};
  const int ar = tid >> 2, ac = (tid & 3) << 3;
  const int br = tid >> 3, bc = (tid & 7) << 3;
  const int gr = m0 + ar;
  const int bb = gr / NPAT, pp = gr % NPAT;
  const int py = pp / 10, px = pp % 10;

  for (int kk = 0; kk < DM; kk += 32) {
    {
      const int k = kk + ac;
      const int cch = k >> 8, ii = (k >> 4) & 15, jj = k & 15;
      const float* src =
          x + ((size_t)(bb * 3 + cch) * 256 + (py * 12 + ii)) * 128 + px * 12 + jj;
#pragma unroll
      for (int e = 0; e < 8; ++e) sA[ar * 40 + ac + e] = (bf16)src[e];
      *(v8bf*)(sB + br * 72 + bc) =
          *(const v8bf*)(WcT + (size_t)(kk + br) * DM + n0 + bc);
    }
    __syncthreads();
    {
      const bf16* ap = sA + (wr * 16 + (lane & 15)) * 40 + ((lane >> 4) << 3);
      v8bf alo = *(const v8bf*)ap;
      v8bf ahi = *(const v8bf*)(ap + 16);
      v16bf af;
#pragma unroll
      for (int i = 0; i < 8; ++i) { af[i] = alo[i]; af[i + 8] = ahi[i]; }
      const int kb = (lane >> 4) << 3;
      const bf16* bp = sB + wc * 32 + (lane & 15);
      v16bf bf0, bf1;
#pragma unroll
      for (int i = 0; i < 8; ++i) {
        bf0[i] = bp[(kb + i) * 72]; bf0[i + 8] = bp[(kb + 16 + i) * 72];
        bf1[i] = bp[(kb + i) * 72 + 16]; bf1[i + 8] = bp[(kb + 16 + i) * 72 + 16];
      }
      acc0 = wmma_bf(af, bf0, acc0);
      acc1 = wmma_bf(af, bf1, acc1);
    }
    __syncthreads();
  }
  const int rbase = wr * 16 + ((lane >> 4) << 3);
  const int c0 = n0 + wc * 32 + (lane & 15);
#pragma unroll
  for (int j = 0; j < 8; ++j) {
    const int grr = m0 + rbase + j;
    const int b2 = grr / NPAT, p2 = grr % NPAT;
    const int cid = cids[b2];
    float* dst = t + ((size_t)b2 * NSEQ + 1 + p2) * DM;
    dst[c0] = acc0[j] + convb[c0] + pos[(1 + p2) * DM + c0] + 3.0f * sie[cid * DM + c0];
    dst[c0 + 16] = acc1[j] + convb[c0 + 16] + pos[(1 + p2) * DM + c0 + 16] +
                   3.0f * sie[cid * DM + c0 + 16];
  }
}

// ---------------------------------------------------------------- layernorm -> bf16
__global__ __launch_bounds__(256) void k_ln_bf16(const float* __restrict__ x,
                                                 const float* __restrict__ g,
                                                 const float* __restrict__ bt,
                                                 bf16* __restrict__ out) {
  __shared__ float red[8];
  const size_t row = blockIdx.x;
  const float* xr = x + row * DM;
  const int tid = threadIdx.x;
  float v0 = xr[tid], v1 = xr[tid + 256], v2 = xr[tid + 512];
  float s = v0 + v1 + v2;
#pragma unroll
  for (int off = 16; off > 0; off >>= 1) s += __shfl_xor(s, off, 32);
  if ((tid & 31) == 0) red[tid >> 5] = s;
  __syncthreads();
  float tot = 0.f;
#pragma unroll
  for (int i = 0; i < 8; ++i) tot += red[i];
  const float mean = tot * (1.0f / DM);
  __syncthreads();
  const float d0 = v0 - mean, d1 = v1 - mean, d2 = v2 - mean;
  float vs = d0 * d0 + d1 * d1 + d2 * d2;
#pragma unroll
  for (int off = 16; off > 0; off >>= 1) vs += __shfl_xor(vs, off, 32);
  if ((tid & 31) == 0) red[tid >> 5] = vs;
  __syncthreads();
  tot = 0.f;
#pragma unroll
  for (int i = 0; i < 8; ++i) tot += red[i];
  const float inv = rsqrtf(tot * (1.0f / DM) + 1e-6f);
  bf16* orow = out + row * DM;
  orow[tid] = (bf16)(d0 * inv * g[tid] + bt[tid]);
  orow[tid + 256] = (bf16)(d1 * inv * g[tid + 256] + bt[tid + 256]);
  orow[tid + 512] = (bf16)(d2 * inv * g[tid + 512] + bt[tid + 512]);
}

// layernorm of one row per batch -> d_out feature slot (fp32, scaled)
__global__ __launch_bounds__(256) void k_ln_feat(const float* __restrict__ base,
                                                 long rowStride,
                                                 const float* __restrict__ g,
                                                 const float* __restrict__ bt,
                                                 float* __restrict__ out,
                                                 int outOff, float mul) {
  __shared__ float red[8];
  const float* xr = base + (size_t)blockIdx.x * rowStride;
  const int tid = threadIdx.x;
  float v0 = xr[tid], v1 = xr[tid + 256], v2 = xr[tid + 512];
  float s = v0 + v1 + v2;
#pragma unroll
  for (int off = 16; off > 0; off >>= 1) s += __shfl_xor(s, off, 32);
  if ((tid & 31) == 0) red[tid >> 5] = s;
  __syncthreads();
  float tot = 0.f;
#pragma unroll
  for (int i = 0; i < 8; ++i) tot += red[i];
  const float mean = tot * (1.0f / DM);
  __syncthreads();
  const float d0 = v0 - mean, d1 = v1 - mean, d2 = v2 - mean;
  float vs = d0 * d0 + d1 * d1 + d2 * d2;
#pragma unroll
  for (int off = 16; off > 0; off >>= 1) vs += __shfl_xor(vs, off, 32);
  if ((tid & 31) == 0) red[tid >> 5] = vs;
  __syncthreads();
  tot = 0.f;
#pragma unroll
  for (int i = 0; i < 8; ++i) tot += red[i];
  const float inv = rsqrtf(tot * (1.0f / DM) + 1e-6f);
  float* orow = out + (size_t)blockIdx.x * 3840 + outOff;
  orow[tid] = (d0 * inv * g[tid] + bt[tid]) * mul;
  orow[tid + 256] = (d1 * inv * g[tid + 256] + bt[tid + 256]) * mul;
  orow[tid + 512] = (d2 * inv * g[tid + 512] + bt[tid + 512]) * mul;
}

// ---------------------------------------------------------------- attention
// qkv: (Bn*Nn, 2304) bf16  [q|k|v per head]; o: (Bn*Nn, 768) bf16
// 4 waves/block; waves share cooperative async K/V staging, each owns 16 rows.
__global__ __launch_bounds__(128) void k_attn(const bf16* __restrict__ qkv,
                                              bf16* __restrict__ o, int Nn) {
  __shared__ __align__(16) bf16 sK[16 * 72];
  __shared__ __align__(16) bf16 sV[32 * 72];
  __shared__ __align__(16) float sS[4][16 * 224];
  const int tid = threadIdx.x;
  const int lane = tid & 31;
  const int wv = tid >> 5;
  const int r0 = blockIdx.x * 64 + wv * 16;
  const int h = blockIdx.y;
  const int b = blockIdx.z;
  const size_t rowBase = (size_t)b * Nn;
  const int QO = h * HDIM, KO = DM + h * HDIM, VO = 2 * DM + h * HDIM;
  float* mS = sS[wv];

  v16bf qf0, qf1;
  {
    int tok = r0 + (lane & 15);
    if (tok >= Nn) tok = Nn - 1;
    const bf16* qp = qkv + (rowBase + tok) * 2304 + QO + ((lane >> 4) << 3);
    v8bf a0 = *(const v8bf*)qp;
    v8bf a1 = *(const v8bf*)(qp + 16);
    v8bf a2 = *(const v8bf*)(qp + 32);
    v8bf a3 = *(const v8bf*)(qp + 48);
#pragma unroll
    for (int i = 0; i < 8; ++i) {
      qf0[i] = a0[i]; qf0[i + 8] = a1[i];
      qf1[i] = a2[i]; qf1[i + 8] = a3[i];
    }
  }

  const int nT16 = (Nn + 15) >> 4;
  const int Npad = ((Nn + 31) >> 5) << 5;  // <= 224

  // ---- S = (Q*scale) @ K^T, tile by 16 key tokens
  for (int ct = 0; ct < nT16; ++ct) {
    {
      // 128 threads stage 16 tokens x 64 dims (16B per thread), async to LDS
      int tok = ct * 16 + (tid >> 3);
      const int cc = (tid & 7) << 3;
      if (tok >= Nn) tok = Nn - 1;  // clamped; masked columns overwritten later
      async_cp16(&sK[(tid >> 3) * 72 + cc],
                 qkv + (rowBase + tok) * 2304 + KO + cc);
      wait_async_0();
    }
    __syncthreads();
    v8f sacc = {};
    {
      v16bf kf;
      const bf16* kp = sK + (lane & 15) * 72 + ((lane >> 4) << 3);
      v8bf k0 = *(const v8bf*)kp;
      v8bf k1 = *(const v8bf*)(kp + 16);
      v8bf k2 = *(const v8bf*)(kp + 32);
      v8bf k3 = *(const v8bf*)(kp + 48);
#pragma unroll
      for (int e = 0; e < 8; ++e) { kf[e] = k0[e]; kf[e + 8] = k1[e]; }
      sacc = wmma_bf(qf0, kf, sacc);
#pragma unroll
      for (int e = 0; e < 8; ++e) { kf[e] = k2[e]; kf[e + 8] = k3[e]; }
      sacc = wmma_bf(qf1, kf, sacc);
    }
    {
      const int nn = ct * 16 + (lane & 15);
      const int rr = (lane >> 4) << 3;
#pragma unroll
      for (int j = 0; j < 8; ++j) mS[(rr + j) * 224 + nn] = sacc[j] * 0.125f;
    }
    __syncthreads();
  }
  // mask padded key columns, then exact per-row softmax (one lane per row)
  for (int n = Nn + lane; n < Npad; n += 32)
    for (int r = 0; r < 16; ++r) mS[r * 224 + n] = -1e30f;
  if (lane < 16) {
    float mx = -1e30f;
    for (int n = 0; n < Npad; ++n) mx = fmaxf(mx, mS[lane * 224 + n]);
    float sum = 0.f;
    for (int n = 0; n < Npad; ++n) {
      float e = __expf(mS[lane * 224 + n] - mx);
      mS[lane * 224 + n] = e;
      sum += e;
    }
    const float inv = 1.0f / sum;
    for (int n = 0; n < Npad; ++n) mS[lane * 224 + n] *= inv;
  }

  // ---- O = P @ V, tile by 32 value tokens
  v8f oa[4] = {};
  const int nT32 = Npad >> 5;
  for (int kt = 0; kt < nT32; ++kt) {
    {
      // 128 threads stage 32 tokens x 64 dims (2x16B per thread)
      int tok = kt * 32 + (tid >> 2);
      const int cc = (tid & 3) << 4;
      if (tok >= Nn) tok = Nn - 1;  // P is zero there -> contributes nothing
      const bf16* src = qkv + (rowBase + tok) * 2304 + VO + cc;
      bf16* dst = &sV[(tid >> 2) * 72 + cc];
      async_cp16(dst, src);
      async_cp16(dst + 8, src + 8);
      wait_async_0();
    }
    __syncthreads();
    v16bf pf;
    {
      const float* sp = mS + (lane & 15) * 224 + kt * 32 + ((lane >> 4) << 3);
#pragma unroll
      for (int e = 0; e < 8; ++e) {
        pf[e] = (bf16)sp[e];
        pf[e + 8] = (bf16)sp[e + 16];
      }
    }
#pragma unroll
    for (int c = 0; c < 4; ++c) {
      const bf16* vp = sV + ((lane >> 4) << 3) * 72 + c * 16 + (lane & 15);
      v16bf vf;
#pragma unroll
      for (int i = 0; i < 8; ++i) { vf[i] = vp[i * 72]; vf[i + 8] = vp[(16 + i) * 72]; }
      oa[c] = wmma_bf(pf, vf, oa[c]);
    }
    __syncthreads();
  }
  {
    const int rr = (lane >> 4) << 3;
    const int cbase = h * HDIM + (lane & 15);
#pragma unroll
    for (int j = 0; j < 8; ++j) {
      const int row = r0 + rr + j;
      if (row < Nn) {
        bf16* op = o + (rowBase + row) * DM + cbase;
        op[0] = (bf16)oa[0][j];
        op[16] = (bf16)oa[1][j];
        op[32] = (bf16)oa[2][j];
        op[48] = (bf16)oa[3][j];
      }
    }
  }
}

// ---------------------------------------------------------------- seg gather
// seg[(i*64+b)*53 + r] rows per reference shift/group/segment mapping
__global__ __launch_bounds__(256) void k_gather_seg(const float* __restrict__ t,
                                                    float* __restrict__ seg) {
  const int ib = blockIdx.x;
  const int i = ib >> 6, b = ib & 63;
  const int r = blockIdx.y;
  int srcTok;
  if (r == 0) {
    srcTok = 0;
  } else {
    const int jj = i * 52 + (r - 1);
    const int g = jj & 1, rr = jj >> 1;
    const int s = g * 105 + rr;
    srcTok = (s < 206) ? (5 + s) : (1 + (s - 206));
  }
  float* dst = seg + ((size_t)(i * 64 + b) * NSEG + r) * DM;
  const float* src = t + ((size_t)b * NSEQ + srcTok) * DM;
  for (int c = threadIdx.x; c < DM; c += 256) dst[c] = src[c];
}

// ================================================================= host
extern "C" void kernel_launch(void* const* d_in, const int* in_sizes, int n_in,
                              void* d_out, int out_size, void* d_ws, size_t ws_size,
                              hipStream_t stream) {
  (void)in_sizes; (void)n_in; (void)out_size; (void)ws_size;
  const float* x      = (const float*)d_in[0];
  const int*   cids   = (const int*)d_in[1];
  const float* conv_w = (const float*)d_in[2];
  const float* conv_b = (const float*)d_in[3];
  const float* cls_t  = (const float*)d_in[4];
  const float* pos    = (const float*)d_in[5];
  const float* sie    = (const float*)d_in[6];
  const float* blk_ln1_g = (const float*)d_in[7];
  const float* blk_ln1_b = (const float*)d_in[8];
  const float* blk_qkv_w = (const float*)d_in[9];
  const float* blk_qkv_b = (const float*)d_in[10];
  const float* blk_proj_w = (const float*)d_in[11];
  const float* blk_proj_b = (const float*)d_in[12];
  const float* blk_ln2_g = (const float*)d_in[13];
  const float* blk_ln2_b = (const float*)d_in[14];
  const float* blk_fc1_w = (const float*)d_in[15];
  const float* blk_fc1_b = (const float*)d_in[16];
  const float* blk_fc2_w = (const float*)d_in[17];
  const float* blk_fc2_b = (const float*)d_in[18];
  const float* b1p[12];
  for (int i = 0; i < 12; ++i) b1p[i] = (const float*)d_in[19 + i];
  const float* b1_norm_g = (const float*)d_in[31];
  const float* b1_norm_b = (const float*)d_in[32];
  const float* b2p[12];
  for (int i = 0; i < 12; ++i) b2p[i] = (const float*)d_in[33 + i];
  const float* b2_norm_g = (const float*)d_in[45];
  const float* b2_norm_b = (const float*)d_in[46];
  float* out = (float*)d_out;

  // ---- workspace carve
  char* wp = (char*)d_ws;
  auto carve = [&](size_t bytes) -> void* {
    void* r = (void*)wp;
    wp += (bytes + 255) & ~(size_t)255;
    return r;
  };
  bf16* wConvT = (bf16*)carve((size_t)DM * DM * 2);
  bf16* wQkv   = (bf16*)carve((size_t)11 * DM * 3 * DM * 2);
  bf16* wProj  = (bf16*)carve((size_t)11 * DM * DM * 2);
  bf16* wFc1   = (bf16*)carve((size_t)11 * DM * MLPD * 2);
  bf16* wFc2   = (bf16*)carve((size_t)11 * MLPD * DM * 2);
  bf16* b1Qkv = (bf16*)carve((size_t)DM * 3 * DM * 2);
  bf16* b1Proj = (bf16*)carve((size_t)DM * DM * 2);
  bf16* b1Fc1 = (bf16*)carve((size_t)DM * MLPD * 2);
  bf16* b1Fc2 = (bf16*)carve((size_t)MLPD * DM * 2);
  bf16* b2Qkv = (bf16*)carve((size_t)DM * 3 * DM * 2);
  bf16* b2Proj = (bf16*)carve((size_t)DM * DM * 2);
  bf16* b2Fc1 = (bf16*)carve((size_t)DM * MLPD * 2);
  bf16* b2Fc2 = (bf16*)carve((size_t)MLPD * DM * 2);
  float* t   = (float*)carve((size_t)NB * NSEQ * DM * 4);
  float* u   = (float*)carve((size_t)NB * NSEQ * DM * 4);
  float* seg = (float*)carve((size_t)NSEGB * NSEG * DM * 4);
  bf16* h_bf   = (bf16*)carve((size_t)MMAX * DM * 2);
  bf16* qkv_bf = (bf16*)carve((size_t)MMAX * 3 * DM * 2);
  bf16* o_bf   = (bf16*)carve((size_t)MMAX * DM * 2);
  bf16* hid_bf = (bf16*)carve((size_t)MMAX * MLPD * 2);

  // ---- weight conversion (fp32 -> bf16), once per launch
  auto cvt = [&](const float* s, bf16* d, long n) {
    k_cvt<<<dim3(2048), dim3(256), 0, stream>>>(s, d, n);
  };
  k_cvtT<<<dim3(48, 48), dim3(256), 0, stream>>>(conv_w, wConvT);
  cvt(blk_qkv_w, wQkv, 11L * DM * 3 * DM);
  cvt(blk_proj_w, wProj, 11L * DM * DM);
  cvt(blk_fc1_w, wFc1, 11L * DM * MLPD);
  cvt(blk_fc2_w, wFc2, 11L * MLPD * DM);
  cvt(b1p[2], b1Qkv, (long)DM * 3 * DM);
  cvt(b1p[4], b1Proj, (long)DM * DM);
  cvt(b1p[8], b1Fc1, (long)DM * MLPD);
  cvt(b1p[10], b1Fc2, (long)MLPD * DM);
  cvt(b2p[2], b2Qkv, (long)DM * 3 * DM);
  cvt(b2p[4], b2Proj, (long)DM * DM);
  cvt(b2p[8], b2Fc1, (long)DM * MLPD);
  cvt(b2p[10], b2Fc2, (long)MLPD * DM);

  // ---- embeddings
  k_cls<<<dim3(NB), dim3(256), 0, stream>>>(cls_t, pos, sie, cids, t);
  k_patch_embed<<<dim3(DM / 64, NB * NPAT / 64), dim3(256), 0, stream>>>(
      x, wConvT, conv_b, pos, sie, cids, t);

  // ---- one transformer block (pre-norm ViT)
  auto run_block = [&](float* xb, int Bn, int Nn, const float* ln1g,
                       const float* ln1b, const bf16* qw, const float* qb,
                       const bf16* pw, const float* pb, const float* ln2g,
                       const float* ln2b, const bf16* f1w, const float* f1b,
                       const bf16* f2w, const float* f2b) {
    const int M = Bn * Nn;
    const int mb = (M + 63) / 64;
    k_ln_bf16<<<dim3(M), dim3(256), 0, stream>>>(xb, ln1g, ln1b, h_bf);
    k_gemm<3><<<dim3(3 * DM / 64, mb), dim3(256), 0, stream>>>(
        h_bf, qw, qb, nullptr, nullptr, qkv_bf, M, DM, 3 * DM);
    k_attn<<<dim3((Nn + 63) / 64, NHEAD, Bn), dim3(128), 0, stream>>>(qkv_bf, o_bf, Nn);
    k_gemm<1><<<dim3(DM / 64, mb), dim3(256), 0, stream>>>(
        o_bf, pw, pb, xb, xb, nullptr, M, DM, DM);
    k_ln_bf16<<<dim3(M), dim3(256), 0, stream>>>(xb, ln2g, ln2b, h_bf);
    k_gemm<2><<<dim3(MLPD / 64, mb), dim3(256), 0, stream>>>(
        h_bf, f1w, f1b, nullptr, nullptr, hid_bf, M, DM, MLPD);
    k_gemm<1><<<dim3(DM / 64, mb), dim3(256), 0, stream>>>(
        hid_bf, f2w, f2b, xb, xb, nullptr, M, MLPD, DM);
  };

  // ---- 11 stacked blocks
  for (int L = 0; L < 11; ++L) {
    run_block(t, NB, NSEQ, blk_ln1_g + L * DM, blk_ln1_b + L * DM,
              wQkv + (size_t)L * DM * 3 * DM, blk_qkv_b + L * 3 * DM,
              wProj + (size_t)L * DM * DM, blk_proj_b + L * DM,
              blk_ln2_g + L * DM, blk_ln2_b + L * DM,
              wFc1 + (size_t)L * DM * MLPD, blk_fc1_b + L * MLPD,
              wFc2 + (size_t)L * MLPD * DM, blk_fc2_b + L * DM);
  }

  // ---- b1 branch -> f0
  hipMemcpyAsync(u, t, (size_t)NB * NSEQ * DM * 4, hipMemcpyDeviceToDevice, stream);
  run_block(u, NB, NSEQ, b1p[0], b1p[1], b1Qkv, b1p[3], b1Proj, b1p[5], b1p[6],
            b1p[7], b1Fc1, b1p[9], b1Fc2, b1p[11]);
  k_ln_feat<<<dim3(NB), dim3(256), 0, stream>>>(u, (long)NSEQ * DM, b1_norm_g,
                                                b1_norm_b, out, 0, 1.0f);

  // ---- b2 branch: 4 shifted/grouped segments batched as B'=256, N'=53
  k_gather_seg<<<dim3(NSEGB, NSEG), dim3(256), 0, stream>>>(t, seg);
  run_block(seg, NSEGB, NSEG, b2p[0], b2p[1], b2Qkv, b2p[3], b2Proj, b2p[5],
            b2p[6], b2p[7], b2Fc1, b2p[9], b2Fc2, b2p[11]);
  for (int i = 0; i < 4; ++i) {
    k_ln_feat<<<dim3(NB), dim3(256), 0, stream>>>(
        seg + (size_t)i * NB * NSEG * DM, (long)NSEG * DM, b2_norm_g, b2_norm_b,
        out, (1 + i) * DM, 0.25f);
  }
}
